// Criterion_90486370992708
// MI455X (gfx1250) — matrix operand order repaired and checked by reading, not compiled, not run on hardware
//
#include <hip/hip_runtime.h>
#include <hip/hip_fp16.h>
#include <math.h>

typedef __attribute__((ext_vector_type(16))) _Float16 v16h;
typedef __attribute__((ext_vector_type(8)))  _Float16 v8h;
typedef __attribute__((ext_vector_type(8)))  float    v8f;

#define BSZ   4096
#define DDIM  1024
#define POS_W 2.0f
#define NEG_W 40.0f
#define MARG  0.1f
#define EPSV  1e-5f

// ---------------------------------------------------------------- f32 -> f16
__global__ __launch_bounds__(256) void cvt_f16_kernel(
    const float* __restrict__ in, _Float16* __restrict__ out, int n) {
  int i = blockIdx.x * 256 + threadIdx.x;
  if (i < n) out[i] = (_Float16)in[i];
}

// ------------------------------------------------- main: 16-row sim strip/WG
// One workgroup (8 wave32) computes a 16x4096 strip of sim = A*A^T via WMMA,
// holds it in LDS (256KB of the 320KB/WGP), then does both mining passes and
// the masked log-sum-exp entirely on-chip. Each wave register-blocks 4 column
// tiles so the A fragment is loaded once per 4 WMMAs (2.5 b128 loads/WMMA).
__global__ __launch_bounds__(256) void ms_loss_rows_kernel(
    const _Float16* __restrict__ Ah,     // [BSZ][DDIM] f16
    const float*    __restrict__ beta,   // [100]
    const int*      __restrict__ labels, // [BSZ]
    float*          __restrict__ rowLoss) {
  extern __shared__ char smem[];
  float* strip = (float*)smem;                               // 16*4096 f32
  int*   slab  = (int*)  (smem + 16 * BSZ * 4);              // 4096 int
  float* redA  = (float*)(smem + 16 * BSZ * 4 + BSZ * 4);    // 256
  float* redB  = redA + 256;                                 // 256

  const int tid     = threadIdx.x;
  const int lane    = tid & 31;          // wave32 lane
  const int wave    = tid >> 5;          // 0..7
  const int rowBase = blockIdx.x * 16;

  // Cache all labels in LDS (uniform loop, no divergence).
  for (int i = tid; i < BSZ; i += 256) slab[i] = labels[i];

  // ---- Phase 1: WMMA GEMM strip -> LDS --------------------------------
  const int m    = lane & 15;            // row-within-tile / col-within-tile
  const int half = lane >> 4;            // lane group 0/1 (K split)
  // A fragment base: row (rowBase+m), K starts at half*16 (contiguous, matches B)
  const _Float16* aRow = Ah + (size_t)(rowBase + m) * DDIM + half * 16;

  for (int g = 0; g < 8; ++g) {          // 8 groups x 4 column tiles
    const int colBase = (wave * 32 + g * 4) * 16;
    // B fragment: B[k][n] = Ah[colBase+n][k]; lane n holds K = half*16..+15
    const _Float16* bRow0 = Ah + (size_t)(colBase + 0  + m) * DDIM + half * 16;
    const _Float16* bRow1 = Ah + (size_t)(colBase + 16 + m) * DDIM + half * 16;
    const _Float16* bRow2 = Ah + (size_t)(colBase + 32 + m) * DDIM + half * 16;
    const _Float16* bRow3 = Ah + (size_t)(colBase + 48 + m) * DDIM + half * 16;

    v8f acc0 = {}, acc1 = {}, acc2 = {}, acc3 = {};
    #pragma unroll 2
    for (int k0 = 0; k0 < DDIM; k0 += 32) {
      union { v16h v; v8h h[2]; } a, b0, b1, b2, b3;
      a.h[0]  = *(const v8h*)(aRow + k0);        // K = k0 + half*16 + 0..7
      a.h[1]  = *(const v8h*)(aRow + k0 + 8);    // K = k0 + half*16 + 8..15
      b0.h[0] = *(const v8h*)(bRow0 + k0);       // K = k0 + half*16 + 0..7
      b0.h[1] = *(const v8h*)(bRow0 + k0 + 8);   // K = k0 + half*16 + 8..15
      b1.h[0] = *(const v8h*)(bRow1 + k0);
      b1.h[1] = *(const v8h*)(bRow1 + k0 + 8);
      b2.h[0] = *(const v8h*)(bRow2 + k0);
      b2.h[1] = *(const v8h*)(bRow2 + k0 + 8);
      b3.h[0] = *(const v8h*)(bRow3 + k0);
      b3.h[1] = *(const v8h*)(bRow3 + k0 + 8);
      acc0 = __builtin_amdgcn_wmma_f32_16x16x32_f16(false, a.v, false, b0.v,
                                                    (short)0, acc0, false, false);
      acc1 = __builtin_amdgcn_wmma_f32_16x16x32_f16(false, a.v, false, b1.v,
                                                    (short)0, acc1, false, false);
      acc2 = __builtin_amdgcn_wmma_f32_16x16x32_f16(false, a.v, false, b2.v,
                                                    (short)0, acc2, false, false);
      acc3 = __builtin_amdgcn_wmma_f32_16x16x32_f16(false, a.v, false, b3.v,
                                                    (short)0, acc3, false, false);
    }
    // C/D layout: VGPR v -> M = v + 8*half; lane -> N = m
    #pragma unroll
    for (int v = 0; v < 8; ++v) {
      float* dst = strip + (v + 8 * half) * BSZ + colBase + m;
      dst[0]  = acc0[v];
      dst[16] = acc1[v];
      dst[32] = acc2[v];
      dst[48] = acc3[v];
    }
  }
  __syncthreads();

  // ---- Phase 2: row-wise mining + masked LSE on the LDS strip ---------
  const int   r    = tid >> 4;           // row 0..15 (16 threads per row)
  const int   c0   = tid & 15;
  const int   gRow = rowBase + r;
  const int   li   = slab[gRow];
  const float* srow = strip + r * BSZ;

  // Pass 2a: partial min over positives / max over negatives.
  float minp = INFINITY, maxn = -INFINITY;
  for (int c = c0; c < BSZ; c += 16) {
    float s = srow[c];
    if (slab[c] == li) {
      if (c != gRow && s < 1.0f - EPSV) minp = fminf(minp, s);
    } else {
      maxn = fmaxf(maxn, s);
    }
  }
  redA[tid] = minp;
  redB[tid] = maxn;
  __syncthreads();

  float rowMin = INFINITY, rowMax = -INFINITY;   // redundant per-row reduce
  #pragma unroll
  for (int t = 0; t < 16; ++t) {
    rowMin = fminf(rowMin, redA[r * 16 + t]);
    rowMax = fmaxf(rowMax, redB[r * 16 + t]);
  }
  __syncthreads();                               // before reusing redA/redB

  // Pass 2b: masked exp sums (any(keep) == sum > 0 since exp() > 0 here).
  const float bb = beta[li];
  float psum = 0.0f, nsum = 0.0f;
  for (int c = c0; c < BSZ; c += 16) {
    float s = srow[c];
    if (slab[c] == li) {
      if (c != gRow && s < 1.0f - EPSV && (s - MARG < rowMax))
        psum += __expf(-POS_W * (s - bb));
    } else {
      if (s + MARG > rowMin)
        nsum += __expf(NEG_W * (s - bb));
    }
  }
  redA[tid] = psum;
  redB[tid] = nsum;
  __syncthreads();

  if (c0 == 0) {
    float P = 0.0f, N = 0.0f;
    #pragma unroll
    for (int t = 0; t < 16; ++t) { P += redA[r * 16 + t]; N += redB[r * 16 + t]; }
    float loss = 0.0f;
    if (P > 0.0f && N > 0.0f)
      loss = (1.0f / POS_W) * log1pf(P) + (1.0f / NEG_W) * log1pf(N);
    rowLoss[gRow] = loss;
  }
}

// ------------------------------------------- deterministic final reduction
__global__ __launch_bounds__(256) void final_reduce_kernel(
    const float* __restrict__ rowLoss, float* __restrict__ out) {
  __shared__ float sh[256];
  float s = 0.0f;
  for (int i = threadIdx.x; i < BSZ; i += 256) s += rowLoss[i];
  sh[threadIdx.x] = s;
  __syncthreads();
  for (int off = 128; off > 0; off >>= 1) {
    if (threadIdx.x < off) sh[threadIdx.x] += sh[threadIdx.x + off];
    __syncthreads();
  }
  if (threadIdx.x == 0) out[0] = sh[0] / (float)BSZ;
}

extern "C" void kernel_launch(void* const* d_in, const int* in_sizes, int n_in,
                              void* d_out, int out_size, void* d_ws, size_t ws_size,
                              hipStream_t stream) {
  const float* batch  = (const float*)d_in[0];  // [4096][1024] f32
  const float* beta   = (const float*)d_in[1];  // [100] f32
  const int*   labels = (const int*)d_in[2];    // [4096]
  float*       out    = (float*)d_out;          // scalar

  _Float16* Ah      = (_Float16*)d_ws;                                // 8 MB
  float*    rowLoss = (float*)((char*)d_ws + (size_t)BSZ * DDIM * 2); // 16 KB

  const int n = BSZ * DDIM;
  cvt_f16_kernel<<<(n + 255) / 256, 256, 0, stream>>>(batch, Ah, n);

  // LDS: 16x4096 f32 strip + 4096 labels + 2*256 reduction floats
  const size_t shmem = (size_t)16 * BSZ * 4 + (size_t)BSZ * 4 + 512 * 4;
  ms_loss_rows_kernel<<<BSZ / 16, 256, shmem, stream>>>(Ah, beta, labels, rowLoss);

  final_reduce_kernel<<<1, 256, 0, stream>>>(rowLoss, out);
}
